// UNetTransformer_77661598646729
// MI455X (gfx1250) — matrix operand order repaired and checked by reading, not compiled, not run on hardware
//
#include <hip/hip_runtime.h>
#include <stdint.h>

// ---------------------------------------------------------------------------
// Types for CDNA5 WMMA (wave32)
// ---------------------------------------------------------------------------
typedef _Float16 v16h __attribute__((ext_vector_type(16)));
typedef _Float16 v8h  __attribute__((ext_vector_type(8)));
typedef float    v8f  __attribute__((ext_vector_type(8)));

#define TM 128   // block tile M
#define TN 64    // block tile N
#define TKK 64   // K-step: two 16x16x32 WMMA K-slices

// ---------------------------------------------------------------------------
// Generic GEMM: C[M,N] = A[M,K] * W[N,K]^T + bias[N] (+ R[M,N]) (+relu)
// 256 threads = 8 waves laid out 4(M) x 2(N); each wave computes a 32x32
// output block = 2x2 WMMA tiles with 4 independent f32 accumulators.
// Per K-step(64): 8 v_wmma_f32_16x16x32_f16 fed by 16 ds_load_b128, with
// A/B fragments reused across the 2x2 register block.
// f32 inputs staged to LDS as f16 (b128 global loads -> v_cvt_pk_f16_f32 ->
// b128 DS stores).  Requires M%128==0, N%64==0, K%64==0.
// ---------------------------------------------------------------------------
__global__ __launch_bounds__(256)
void gemm_wmma_kernel(const float* __restrict__ A, const float* __restrict__ W,
                      const float* __restrict__ bias, const float* __restrict__ R,
                      float* __restrict__ C, int M, int N, int K, int relu)
{
  __shared__ _Float16 sA[TM][TKK];   // 16 KB
  __shared__ _Float16 sB[TN][TKK];   //  8 KB
  const int t    = threadIdx.x;
  const int l    = t & 31;          // lane in wave32
  const int wave = t >> 5;
  const int wm   = wave & 3;        // M 32-subtile
  const int wn   = wave >> 2;       // N 32-subtile
  const int bm   = blockIdx.y * TM;
  const int bn   = blockIdx.x * TN;

  v8f acc[2][2] = {};

  // Fragment addressing (this code's contiguous K order, same for A and B):
  // A (16x32 f16): lane m=l&15; halves 0..15 -> K = a0..a0+15,
  //                a0 = (l>=16)?16:0
  // B (32x16 f16): lane n=l&15; halves 0..15 -> K = kb..kb+15, kb=(l>=16)?16:0
  const int lm    = l & 15;
  const int abase = (l >= 16) ? 16 : 0;
  const int kbb   = (l >= 16) ? 16 : 0;

  for (int k0 = 0; k0 < K; k0 += TKK) {
    // ---- Stage A tile (128x64 f32 -> f16): 1024 groups of 8 floats, 4/thread
    float4 a0[4], a1[4];
#pragma unroll
    for (int i = 0; i < 4; ++i) {
      int lin = i * 256 + t;
      int r = lin >> 3, g = lin & 7;
      const float4* src = (const float4*)&A[(size_t)(bm + r) * K + k0 + g * 8];
      a0[i] = src[0];
      a1[i] = src[1];
    }
    // ---- Stage B tile (64x64 f32 -> f16): 512 groups of 8 floats, 2/thread
    float4 b0g[2], b1g[2];
#pragma unroll
    for (int i = 0; i < 2; ++i) {
      int lin = i * 256 + t;
      int r = lin >> 3, g = lin & 7;
      const float4* src = (const float4*)&W[(size_t)(bn + r) * K + k0 + g * 8];
      b0g[i] = src[0];
      b1g[i] = src[1];
    }
#pragma unroll
    for (int i = 0; i < 4; ++i) {
      int lin = i * 256 + t;
      int r = lin >> 3, g = lin & 7;
      v8h h;
      h[0] = (_Float16)a0[i].x; h[1] = (_Float16)a0[i].y;
      h[2] = (_Float16)a0[i].z; h[3] = (_Float16)a0[i].w;
      h[4] = (_Float16)a1[i].x; h[5] = (_Float16)a1[i].y;
      h[6] = (_Float16)a1[i].z; h[7] = (_Float16)a1[i].w;
      *(v8h*)&sA[r][g * 8] = h;
    }
#pragma unroll
    for (int i = 0; i < 2; ++i) {
      int lin = i * 256 + t;
      int r = lin >> 3, g = lin & 7;
      v8h h;
      h[0] = (_Float16)b0g[i].x; h[1] = (_Float16)b0g[i].y;
      h[2] = (_Float16)b0g[i].z; h[3] = (_Float16)b0g[i].w;
      h[4] = (_Float16)b1g[i].x; h[5] = (_Float16)b1g[i].y;
      h[6] = (_Float16)b1g[i].z; h[7] = (_Float16)b1g[i].w;
      *(v8h*)&sB[r][g * 8] = h;
    }
    if (k0 + TKK < K) {
      // warm L2/WGP$ for the next K-tile (global_prefetch_b8 path)
      __builtin_prefetch(&A[(size_t)(bm + (t >> 1)) * K + (k0 + TKK) + ((t & 1) << 5)], 0, 1);
    }
    __syncthreads();

#pragma unroll
    for (int kk = 0; kk < TKK; kk += 32) {
      v16h fa[2], fb[2];
#pragma unroll
      for (int mi = 0; mi < 2; ++mi) {
        const int ml = wm * 32 + mi * 16 + lm;
        v8h lo = *(const v8h*)&sA[ml][kk + abase];
        v8h hi = *(const v8h*)&sA[ml][kk + abase + 8];
#pragma unroll
        for (int i = 0; i < 8; ++i) { fa[mi][i] = lo[i]; fa[mi][i + 8] = hi[i]; }
      }
#pragma unroll
      for (int ni = 0; ni < 2; ++ni) {
        const int nl = wn * 32 + ni * 16 + lm;
        v8h lo = *(const v8h*)&sB[nl][kk + kbb];
        v8h hi = *(const v8h*)&sB[nl][kk + kbb + 8];
#pragma unroll
        for (int i = 0; i < 8; ++i) { fb[ni][i] = lo[i]; fb[ni][i + 8] = hi[i]; }
      }
#pragma unroll
      for (int mi = 0; mi < 2; ++mi)
#pragma unroll
        for (int ni = 0; ni < 2; ++ni)
          acc[mi][ni] = __builtin_amdgcn_wmma_f32_16x16x32_f16(
              false, fa[mi], false, fb[ni], (short)0, acc[mi][ni], false, false);
    }
    __syncthreads();
  }

  // C/D layout: lanes 0-15 N=lane, M=vgpr; lanes 16-31 N=lane-16, M=vgpr+8
#pragma unroll
  for (int mi = 0; mi < 2; ++mi) {
#pragma unroll
    for (int ni = 0; ni < 2; ++ni) {
      const int ngl = bn + wn * 32 + ni * 16 + lm;
      const int mb  = bm + wm * 32 + mi * 16 + ((l >= 16) ? 8 : 0);
      const float bval = bias ? bias[ngl] : 0.0f;
#pragma unroll
      for (int r2 = 0; r2 < 8; ++r2) {
        int m = mb + r2;
        float v = acc[mi][ni][r2] + bval;
        if (R) v += R[(size_t)m * N + ngl];
        if (relu) v = fmaxf(v, 0.0f);
        C[(size_t)m * N + ngl] = v;
      }
    }
  }
}

// ---------------------------------------------------------------------------
// InstanceNorm1d over points axis per (b,c) + ReLU, in-place.
// grid = B * (C/32); block 256 = 32 channels x 8 rows
// ---------------------------------------------------------------------------
__global__ __launch_bounds__(256)
void inorm_relu_kernel(float* __restrict__ x, int Np, int Cc)
{
  __shared__ float ssum[256], ssq[256], sm[32], sr[32];
  const int cblocks = Cc >> 5;
  const int b  = blockIdx.x / cblocks;
  const int cb = (blockIdx.x % cblocks) << 5;
  const int t = threadIdx.x, tx = t & 31, ty = t >> 5;
  const int c = cb + tx;
  float s = 0.f, ss = 0.f;
  for (int n = ty; n < Np; n += 8) {
    float v = x[((size_t)b * Np + n) * Cc + c];
    s += v; ss += v * v;
  }
  ssum[t] = s; ssq[t] = ss;
  __syncthreads();
  for (int off = 128; off >= 32; off >>= 1) {
    if (t < off) { ssum[t] += ssum[t + off]; ssq[t] += ssq[t + off]; }
    __syncthreads();
  }
  if (t < 32) {
    float m = ssum[t] / (float)Np;
    float var = ssq[t] / (float)Np - m * m;
    sm[t] = m; sr[t] = rsqrtf(var + 1e-5f);
  }
  __syncthreads();
  const float m = sm[tx], r = sr[tx];
  for (int n = ty; n < Np; n += 8) {
    size_t o = ((size_t)b * Np + n) * Cc + c;
    x[o] = fmaxf((x[o] - m) * r, 0.f);
  }
}

// ---------------------------------------------------------------------------
// LayerNorm over last dim: y = (x-m)*rstd*g + b ; one row per block
// ---------------------------------------------------------------------------
__global__ __launch_bounds__(256)
void layernorm_kernel(const float* __restrict__ x, const float* __restrict__ g,
                      const float* __restrict__ bta, float* __restrict__ y, int d)
{
  __shared__ float ssum[256], ssq[256], smv[2];
  const size_t row = blockIdx.x;
  const int t = threadIdx.x;
  float s = 0.f, ss = 0.f;
  for (int c = t; c < d; c += 256) { float v = x[row * d + c]; s += v; ss += v * v; }
  ssum[t] = s; ssq[t] = ss;
  __syncthreads();
  for (int off = 128; off >= 1; off >>= 1) {
    if (t < off) { ssum[t] += ssum[t + off]; ssq[t] += ssq[t + off]; }
    __syncthreads();
  }
  if (t == 0) {
    float m = ssum[0] / (float)d;
    smv[0] = m; smv[1] = rsqrtf(ssq[0] / (float)d - m * m + 1e-5f);
  }
  __syncthreads();
  const float m = smv[0], r = smv[1];
  for (int c = t; c < d; c += 256)
    y[row * d + c] = (x[row * d + c] - m) * r * g[c] + bta[c];
}

// ---------------------------------------------------------------------------
// Farthest point sampling (is_center start). One workgroup per batch.
// Running min-distance array lives in global workspace (distg).
// ---------------------------------------------------------------------------
__global__ __launch_bounds__(256)
void fps_kernel(const float* __restrict__ pts, int* __restrict__ idx,
                float* __restrict__ distg, int N, int np)
{
  __shared__ float red[256];
  __shared__ int   redi[256];
  __shared__ float cur[3];
  const int b = blockIdx.x, t = threadIdx.x;
  const float* P = pts + (size_t)b * N * 3;
  float* dist = distg + (size_t)b * N;

  // centroid
  float sx = 0.f, sy = 0.f, sz = 0.f;
  for (int n = t; n < N; n += 256) { sx += P[3*n]; sy += P[3*n+1]; sz += P[3*n+2]; }
  for (int comp = 0; comp < 3; ++comp) {
    red[t] = (comp == 0) ? sx : ((comp == 1) ? sy : sz);
    __syncthreads();
    for (int off = 128; off >= 1; off >>= 1) { if (t < off) red[t] += red[t + off]; __syncthreads(); }
    if (t == 0) cur[comp] = red[0] / (float)N;
    __syncthreads();
  }
  float c0 = cur[0], c1 = cur[1], c2 = cur[2];
  float bmax = -1e30f; int bidx = 0;
  for (int n = t; n < N; n += 256) {
    float dx = P[3*n] - c0, dy = P[3*n+1] - c1, dz = P[3*n+2] - c2;
    float d2 = dx*dx + dy*dy + dz*dz;
    dist[n] = 1e10f;
    if (d2 > bmax) { bmax = d2; bidx = n; }
  }
  red[t] = bmax; redi[t] = bidx;
  __syncthreads();
  for (int off = 128; off >= 1; off >>= 1) {
    if (t < off) {
      if (red[t+off] > red[t] || (red[t+off] == red[t] && redi[t+off] < redi[t])) {
        red[t] = red[t+off]; redi[t] = redi[t+off];
      }
    }
    __syncthreads();
  }
  for (int it = 0; it < np; ++it) {
    if (t == 0) {
      int f = redi[0];
      idx[(size_t)b * np + it] = f;
      cur[0] = P[3*f]; cur[1] = P[3*f+1]; cur[2] = P[3*f+2];
    }
    __syncthreads();
    c0 = cur[0]; c1 = cur[1]; c2 = cur[2];
    bmax = -1e30f; bidx = 0;
    for (int n = t; n < N; n += 256) {
      float dx = P[3*n] - c0, dy = P[3*n+1] - c1, dz = P[3*n+2] - c2;
      float d2 = dx*dx + dy*dy + dz*dz;
      float nd = fminf(dist[n], d2);
      dist[n] = nd;
      if (nd > bmax) { bmax = nd; bidx = n; }
    }
    __syncthreads();
    red[t] = bmax; redi[t] = bidx;
    __syncthreads();
    for (int off = 128; off >= 1; off >>= 1) {
      if (t < off) {
        if (red[t+off] > red[t] || (red[t+off] == red[t] && redi[t+off] < redi[t])) {
          red[t] = red[t+off]; redi[t] = redi[t+off];
        }
      }
      __syncthreads();
    }
  }
}

// ---------------------------------------------------------------------------
// index_points gather: out[b,i,c] = in[b, idx[b,i], c]
// ---------------------------------------------------------------------------
__global__ void gather_kernel(const float* __restrict__ in, const int* __restrict__ idx,
                              float* __restrict__ out, int Nin, int Nout, int C, int total)
{
  int i = blockIdx.x * 256 + threadIdx.x;
  if (i >= total) return;
  int c = i % C; int rest = i / C;
  int n = rest % Nout; int b = rest / Nout;
  out[i] = in[((size_t)b * Nin + idx[(size_t)b * Nout + n]) * C + c];
}

// ---------------------------------------------------------------------------
// Attention over sequence axis S=4 (torch batch_first=False quirk):
// one thread per (bt, head). qkv: [S,Bt,3d] row-major with M = s*Bt+bt.
// ---------------------------------------------------------------------------
__global__ void attn4_kernel(const float* __restrict__ qkv, float* __restrict__ o,
                             int Bt, int d, int total)
{
  int t = blockIdx.x * 256 + threadIdx.x;
  if (t >= total) return;
  const int H = 8, S = 4;
  int bt = t / H, h = t % H, dh = d / H;
  const size_t stride = (size_t)Bt * 3 * d;
  const float* base = qkv + (size_t)bt * 3 * d + h * dh;
  const float scale = rsqrtf((float)dh);
  float sc[4][4];
  for (int q = 0; q < S; ++q)
    for (int k = 0; k < S; ++k) {
      const float* qp = base + (size_t)q * stride;
      const float* kp = base + (size_t)k * stride + d;
      float acc = 0.f;
      for (int e = 0; e < dh; ++e) acc += qp[e] * kp[e];
      sc[q][k] = acc * scale;
    }
  for (int q = 0; q < S; ++q) {
    float mx = sc[q][0];
    for (int k = 1; k < S; ++k) mx = fmaxf(mx, sc[q][k]);
    float sum = 0.f;
    for (int k = 0; k < S; ++k) { sc[q][k] = __expf(sc[q][k] - mx); sum += sc[q][k]; }
    float inv = 1.f / sum;
    for (int k = 0; k < S; ++k) sc[q][k] *= inv;
  }
  for (int q = 0; q < S; ++q) {
    float* op = o + ((size_t)q * Bt + bt) * d + h * dh;
    for (int e = 0; e < dh; ++e) {
      float acc = 0.f;
      for (int k = 0; k < S; ++k) acc += sc[q][k] * base[(size_t)k * stride + 2 * d + e];
      op[e] = acc;
    }
  }
}

// ---------------------------------------------------------------------------
// 3-NN (squared distances) + normalized inverse-distance weights
// one thread per (b, n1)
// ---------------------------------------------------------------------------
__global__ void knn3_kernel(const float* __restrict__ xyz1, const float* __restrict__ xyz2,
                            int* __restrict__ nidx, float* __restrict__ nw,
                            int N1, int N2, int total)
{
  int i = blockIdx.x * 256 + threadIdx.x;
  if (i >= total) return;
  int b = i / N1;
  const float* p = xyz1 + (size_t)i * 3;
  const float px = p[0], py = p[1], pz = p[2];
  const float* q = xyz2 + (size_t)b * N2 * 3;
  float d0 = 1e30f, d1 = 1e30f, d2 = 1e30f;
  int i0 = 0, i1 = 0, i2 = 0;
  for (int j = 0; j < N2; ++j) {
    float dx = px - q[3*j], dy = py - q[3*j+1], dz = pz - q[3*j+2];
    float dd = dx*dx + dy*dy + dz*dz;
    if (dd < d0)      { d2=d1; i2=i1; d1=d0; i1=i0; d0=dd; i0=j; }
    else if (dd < d1) { d2=d1; i2=i1; d1=dd; i1=j; }
    else if (dd < d2) { d2=dd; i2=j; }
  }
  float w0 = 1.f/(d0+1e-8f), w1 = 1.f/(d1+1e-8f), w2 = 1.f/(d2+1e-8f);
  float inv = 1.f/(w0+w1+w2);
  nidx[(size_t)i*3]=i0; nidx[(size_t)i*3+1]=i1; nidx[(size_t)i*3+2]=i2;
  nw[(size_t)i*3]=w0*inv; nw[(size_t)i*3+1]=w1*inv; nw[(size_t)i*3+2]=w2*inv;
}

// ---------------------------------------------------------------------------
// concat [f1 | interp(f2)] -> out[b,n, C1+C2]
// ---------------------------------------------------------------------------
__global__ void interp_concat_kernel(const float* __restrict__ f1, const float* __restrict__ f2,
                                     const int* __restrict__ nidx, const float* __restrict__ nw,
                                     float* __restrict__ out, int N1, int N2, int C1, int C2,
                                     int total)
{
  int i = blockIdx.x * 256 + threadIdx.x;
  if (i >= total) return;
  const int Ct = C1 + C2;
  int c = i % Ct; int rest = i / Ct;
  int n = rest % N1; int b = rest / N1;
  float v;
  if (c < C1) {
    v = f1[((size_t)b * N1 + n) * C1 + c];
  } else {
    int cc = c - C1;
    size_t r = ((size_t)b * N1 + n) * 3;
    const float* fb = f2 + (size_t)b * N2 * C2;
    v = nw[r]     * fb[(size_t)nidx[r]     * C2 + cc]
      + nw[r + 1] * fb[(size_t)nidx[r + 1] * C2 + cc]
      + nw[r + 2] * fb[(size_t)nidx[r + 2] * C2 + cc];
  }
  out[i] = v;
}

// ---------------------------------------------------------------------------
// (B,N,C) -> (B,C,N)
// ---------------------------------------------------------------------------
__global__ void transpose_out_kernel(const float* __restrict__ y, float* __restrict__ out,
                                     int Np, int Cc, int total)
{
  int i = blockIdx.x * 256 + threadIdx.x;
  if (i >= total) return;
  int n = i % Np; int rest = i / Np;
  int c = rest % Cc; int b = rest / Cc;
  out[i] = y[((size_t)b * Np + n) * Cc + c];
}

// ---------------------------------------------------------------------------
// Host-side orchestration
// ---------------------------------------------------------------------------
struct EncP {
  const float *w_in, *b_in, *w_out, *b_out;
  const float *ln1g, *ln1b, *ln2g, *ln2b;
  const float *w1, *b1, *w2, *b2;
};

static EncP enc_params(void* const* d_in, int base) {
  EncP p;
  p.b1    = (const float*)d_in[base + 0];
  p.b2    = (const float*)d_in[base + 1];
  p.b_in  = (const float*)d_in[base + 2];
  p.b_out = (const float*)d_in[base + 3];
  p.ln1b  = (const float*)d_in[base + 4];
  p.ln1g  = (const float*)d_in[base + 5];
  p.ln2b  = (const float*)d_in[base + 6];
  p.ln2g  = (const float*)d_in[base + 7];
  p.w1    = (const float*)d_in[base + 8];
  p.w2    = (const float*)d_in[base + 9];
  p.w_in  = (const float*)d_in[base + 10];
  p.w_out = (const float*)d_in[base + 11];
  return p;
}

static void gemm(hipStream_t s, const float* A, const float* W, const float* b,
                 const float* R, float* C, int M, int N, int K, int relu) {
  dim3 g(N / TN, M / TM);
  gemm_wmma_kernel<<<g, 256, 0, s>>>(A, W, b, R, C, M, N, K, relu);
}

// Post-norm transformer encoder layer (eval). xin may alias x1/xout (x1 is
// written only after the w_out residual read; xout is written only after x1
// is dead).  Hbuf holds qkv + attn output, then the chunked FF hidden.
static void run_encoder(hipStream_t s, const EncP& p, const float* xin, float* xout,
                        float* tmpA, float* x1, float* Hbuf, int Bt, int d) {
  const int M = 4 * Bt;
  float* qkv = Hbuf;
  float* o   = Hbuf + (size_t)M * 3 * d;
  gemm(s, xin, p.w_in, p.b_in, nullptr, qkv, M, 3 * d, d, 0);
  int th = Bt * 8;
  attn4_kernel<<<(th + 255) / 256, 256, 0, s>>>(qkv, o, Bt, d, th);
  gemm(s, o, p.w_out, p.b_out, xin, tmpA, M, d, d, 0);
  layernorm_kernel<<<M, 256, 0, s>>>(tmpA, p.ln1g, p.ln1b, x1, d);
  const int CH = (M >= 4096) ? 4096 : M;
  for (int r0 = 0; r0 < M; r0 += CH) {
    gemm(s, x1 + (size_t)r0 * d, p.w1, p.b1, nullptr, Hbuf, CH, 2048, d, 1);
    gemm(s, Hbuf, p.w2, p.b2, x1 + (size_t)r0 * d, tmpA + (size_t)r0 * d, CH, d, 2048, 0);
  }
  layernorm_kernel<<<M, 256, 0, s>>>(tmpA, p.ln2g, p.ln2b, xout, d);
}

extern "C" void kernel_launch(void* const* d_in, const int* in_sizes, int n_in,
                              void* d_out, int out_size, void* d_ws, size_t ws_size,
                              hipStream_t stream) {
  (void)in_sizes; (void)n_in; (void)out_size; (void)ws_size;

  // Input order: points, features, then params leaves (sorted dict keys):
  // conv0_b, conv0_w, dec[0..2]x12, ds[0..2]x{b,w}, enc[0..2]x12,
  // final_b, final_w, up[0..2]x{b,w}
  const float* points   = (const float*)d_in[0];
  const float* features = (const float*)d_in[1];
  const float* conv0_b  = (const float*)d_in[2];
  const float* conv0_w  = (const float*)d_in[3];
  EncP dec2 = enc_params(d_in, 4 + 24);   // dec[2]
  const float* ds0_b = (const float*)d_in[40];
  const float* ds0_w = (const float*)d_in[41];
  const float* ds1_b = (const float*)d_in[42];
  const float* ds1_w = (const float*)d_in[43];
  const float* ds2_b = (const float*)d_in[44];
  const float* ds2_w = (const float*)d_in[45];
  EncP enc0 = enc_params(d_in, 46);
  EncP enc1 = enc_params(d_in, 58);       // enc[2] is dead code
  const float* final_b = (const float*)d_in[82];
  const float* final_w = (const float*)d_in[83];
  const float* up2_b   = (const float*)d_in[88];
  const float* up2_w   = (const float*)d_in[89];

  // Workspace layout
  uint8_t* wp = (uint8_t*)d_ws;
  auto alloc = [&](size_t bytes) -> void* {
    void* p = (void*)wp;
    wp += (bytes + 255) & ~(size_t)255;
    return p;
  };
  float* F0  = (float*)alloc((size_t)65536 * 128 * 4);   // ping (max 65536x128)
  float* F1  = (float*)alloc((size_t)65536 * 128 * 4);   // pong
  float* HB  = (float*)alloc((size_t)4096 * 2048 * 4);   // qkv+o / ff-hidden
  float* S0F = (float*)alloc((size_t)4 * 4096 * 128 * 4);// skip0 features
  float* S2F = (float*)alloc((size_t)4 * 256 * 512 * 4); // skip2 features
  float* SP0 = (float*)alloc((size_t)4 * 4096 * 3 * 4);  // pts @4096
  float* SP1 = (float*)alloc((size_t)4 * 1024 * 3 * 4);  // pts @1024
  float* SP2 = (float*)alloc((size_t)4 * 256 * 3 * 4);   // pts @256
  int*   IDX = (int*)  alloc((size_t)4 * 4096 * 4);      // fps indices
  int*   NNI = (int*)  alloc((size_t)4 * 4096 * 3 * 4);  // 3-NN indices
  float* NNW = (float*)alloc((size_t)4 * 4096 * 3 * 4);  // 3-NN weights
  float* CAT = (float*)alloc((size_t)4 * 4096 * 640 * 4);// concat buffer
  float* DST = (float*)alloc((size_t)4 * 16384 * 4);     // fps running dist

  // ---- conv0: (4*16384,128) x (128,128)^T
  gemm(stream, features, conv0_w, conv0_b, nullptr, F0, 65536, 128, 128, 0);

  // ---- stage 0: ds0 + inorm + relu
  gemm(stream, F0, ds0_w, ds0_b, nullptr, F1, 65536, 128, 128, 0);
  inorm_relu_kernel<<<4 * (128 / 32), 256, 0, stream>>>(F1, 16384, 128);
  fps_kernel<<<4, 256, 0, stream>>>(points, IDX, DST, 16384, 4096);
  gather_kernel<<<(4 * 4096 * 3 + 255) / 256, 256, 0, stream>>>(points, IDX, SP0, 16384, 4096, 3, 4 * 4096 * 3);
  gather_kernel<<<(4 * 4096 * 128 + 255) / 256, 256, 0, stream>>>(F1, IDX, S0F, 16384, 4096, 128, 4 * 4096 * 128);
  run_encoder(stream, enc0, S0F, /*xout*/F0, /*tmpA*/F1, /*x1*/F0, HB, 4096, 128);

  // ---- stage 1: ds1 + inorm + relu
  gemm(stream, F0, ds1_w, ds1_b, nullptr, F1, 16384, 256, 128, 0);
  inorm_relu_kernel<<<4 * (256 / 32), 256, 0, stream>>>(F1, 4096, 256);
  fps_kernel<<<4, 256, 0, stream>>>(SP0, IDX, DST, 4096, 1024);
  gather_kernel<<<(4 * 1024 * 3 + 255) / 256, 256, 0, stream>>>(SP0, IDX, SP1, 4096, 1024, 3, 4 * 1024 * 3);
  gather_kernel<<<(4 * 1024 * 256 + 255) / 256, 256, 0, stream>>>(F1, IDX, F0, 4096, 1024, 256, 4 * 1024 * 256);
  run_encoder(stream, enc1, F0, /*xout*/F0, /*tmpA*/F1, /*x1*/F0, HB, 1024, 256);

  // ---- stage 2: ds2 + inorm + relu (enc[2] is dead code)
  gemm(stream, F0, ds2_w, ds2_b, nullptr, F1, 4096, 512, 256, 0);
  inorm_relu_kernel<<<4 * (512 / 32), 256, 0, stream>>>(F1, 1024, 512);
  fps_kernel<<<4, 256, 0, stream>>>(SP1, IDX, DST, 1024, 256);
  gather_kernel<<<(4 * 256 * 3 + 255) / 256, 256, 0, stream>>>(SP1, IDX, SP2, 1024, 256, 3, 4 * 256 * 3);
  gather_kernel<<<(4 * 256 * 512 + 255) / 256, 256, 0, stream>>>(F1, IDX, S2F, 1024, 256, 512, 4 * 256 * 512);

  // ---- decode (only the live final iteration): skips[0] <- interp(skips[2])
  knn3_kernel<<<(4 * 4096 + 255) / 256, 256, 0, stream>>>(SP0, SP2, NNI, NNW, 4096, 256, 4 * 4096);
  interp_concat_kernel<<<(4 * 4096 * 640 + 255) / 256, 256, 0, stream>>>(
      S0F, S2F, NNI, NNW, CAT, 4096, 256, 128, 512, 4 * 4096 * 640);
  gemm(stream, CAT, up2_w, up2_b, nullptr, F0, 16384, 128, 640, 0);
  inorm_relu_kernel<<<4 * (128 / 32), 256, 0, stream>>>(F0, 4096, 128);
  run_encoder(stream, dec2, F0, /*xout*/F0, /*tmpA*/F1, /*x1*/F0, HB, 4096, 128);

  // ---- final linear + transpose to (B, OUT_CH, N)
  gemm(stream, F0, final_w, final_b, nullptr, F1, 16384, 64, 128, 0);
  transpose_out_kernel<<<(4 * 64 * 4096 + 255) / 256, 256, 0, stream>>>(
      F1, (float*)d_out, 4096, 64, 4 * 64 * 4096);
}
